// CRF_77232101917010
// MI455X (gfx1250) — compile-verified, matrix operand order checked by reading, not run on hardware
//
#include <hip/hip_runtime.h>

#define B_   8
#define S_   32
#define T_   2048
#define D_   256
#define BEAM_ 5

typedef __bf16 v16bf __attribute__((ext_vector_type(16)));
typedef __bf16 v8bf  __attribute__((ext_vector_type(8)));
typedef float  v8f   __attribute__((ext_vector_type(8)));

// float -> bf16 bits with round-to-nearest-even
__device__ __forceinline__ unsigned short f2bf(float f) {
    unsigned u = __builtin_bit_cast(unsigned, f);
    unsigned r = (u + 0x7FFFu + ((u >> 16) & 1u)) >> 16;
    return (unsigned short)r;
}

// Load a 16-bit A/B WMMA fragment for this lane.
// Per the CDNA5 16-bit 16x32 layout, lane (row m = lane&15, half kh = lane>>4)
// holds K = [kk+8*kh, kk+8*kh+8) in elements 0..7 and
//        K = [kk+16+8*kh, kk+24+8*kh) in elements 8..15
// -> exactly two contiguous 16-byte loads (global_load_b128).
__device__ __forceinline__ v16bf load_frag16(const unsigned short* __restrict__ row,
                                             int kk, int kh) {
    const v8bf* p0 = (const v8bf*)(row + kk + 8 * kh);
    const v8bf* p1 = (const v8bf*)(row + kk + 16 + 8 * kh);
    v8bf lo = *p0, hi = *p1;
    return __builtin_shufflevector(lo, hi, 0, 1, 2, 3, 4, 5, 6, 7,
                                           8, 9, 10, 11, 12, 13, 14, 15);
}

// ---------------------------------------------------------------------------
// Kernel 0: one-shot convert E (f32) -> Ebf (bf16) so the trans GEMM can use
// pure b128 fragment loads.
// ---------------------------------------------------------------------------
__global__ void ebf_kernel(const float* __restrict__ E,
                           unsigned short* __restrict__ Ebf) {
    int idx = blockIdx.x * 256 + threadIdx.x;   // T_*D_ = 524288 threads
    Ebf[idx] = f2bf(E[idx]);
}

// ---------------------------------------------------------------------------
// Kernel 1: P[j,k] = exp( relu( A[j,k] * (E E^T)[j,k] ) )   (bf16 output)
// One 16x16 tile per wave, K-loop over D=256 in chunks of 32 (8 WMMAs).
// ---------------------------------------------------------------------------
__global__ void __launch_bounds__(256) trans_exp_kernel(
        const unsigned short* __restrict__ Ebf, const float* __restrict__ A,
        unsigned short* __restrict__ P) {
    int wave = threadIdx.x >> 5;
    int lane = threadIdx.x & 31;
    int tile = blockIdx.x * 8 + wave;       // 16384 tiles total
    int ti = tile >> 7;                     // row tile (128 tiles)
    int tj = tile & 127;                    // col tile
    int m  = lane & 15;
    int kh = lane >> 4;
    const unsigned short* arow = Ebf + (ti * 16 + m) * D_;
    const unsigned short* brow = Ebf + (tj * 16 + m) * D_;  // B col n == lane&15
    v8f acc = {};
    for (int kk = 0; kk < D_; kk += 32) {
        v16bf a = load_frag16(arow, kk, kh);
        v16bf b = load_frag16(brow, kk, kh);
        acc = __builtin_amdgcn_wmma_f32_16x16x32_bf16(
                false, a, false, b, (short)0, acc, false, false);
    }
#pragma unroll
    for (int r = 0; r < 8; ++r) {
        int gi = ti * 16 + r + 8 * kh;      // D layout: M = r + 8*(lane>=16)
        int gj = tj * 16 + (lane & 15);     //           N = lane & 15
        float t = A[gi * T_ + gj] * acc[r];
        t = t > 0.f ? t : 0.f;
        P[gi * T_ + gj] = f2bf(__expf(t));
    }
}

// ---------------------------------------------------------------------------
// Kernel 2: score[b,:] = emissions[b,0,:]; zero padding rows 8..15 of Q.
// ---------------------------------------------------------------------------
__global__ void init_kernel(const float* __restrict__ emissions,
                            float* __restrict__ score,
                            unsigned short* __restrict__ q) {
    int idx = blockIdx.x * 256 + threadIdx.x;   // 16384 threads
    int b = idx >> 11, j = idx & 2047;
    score[idx] = emissions[b * S_ * T_ + j];
    q[(8 + b) * T_ + j] = 0;                    // +0.0 bf16
}

// ---------------------------------------------------------------------------
// Kernel 3: numerator[b] (gathered emission + relu(dot)*A transition scores)
// One block per batch, 256 threads == D.
// ---------------------------------------------------------------------------
__global__ void __launch_bounds__(256) numer_kernel(
        const float* __restrict__ em, const int* __restrict__ tags,
        const float* __restrict__ E, const float* __restrict__ A,
        const unsigned char* __restrict__ mask, float* __restrict__ num) {
    __shared__ float red[256];
    int b = blockIdx.x, tid = threadIdx.x;
    float acc = 0.f;
    if (tid == 0) acc = em[b * S_ * T_ + tags[b * S_]];
    for (int s = 1; s < S_; ++s) {
        int prev = tags[b * S_ + s - 1];
        int cur  = tags[b * S_ + s];
        red[tid] = E[prev * D_ + tid] * E[cur * D_ + tid];
        __syncthreads();
        for (int off = 128; off > 0; off >>= 1) {
            if (tid < off) red[tid] += red[tid + off];
            __syncthreads();
        }
        if (tid == 0 && mask[b * S_ + s]) {
            float r = red[0] > 0.f ? red[0] : 0.f;
            acc += A[prev * T_ + cur] * r + em[b * S_ * T_ + s * T_ + cur];
        }
        __syncthreads();
    }
    if (tid == 0) num[b] = acc;
}

// ---------------------------------------------------------------------------
// Kernel 4 (per step): row max mb[b], Q[b,k] = exp(score-mb) (bf16),
// top-5 indices via 5 iterated argmax reductions. One block per batch.
// ---------------------------------------------------------------------------
__global__ void __launch_bounds__(256) prep_kernel(
        const float* __restrict__ score, float* __restrict__ mb,
        int* __restrict__ idx5, unsigned short* __restrict__ q) {
    __shared__ float ssc[T_];
    __shared__ float rv[256];
    __shared__ int   ri[256];
    __shared__ float smax;
    int b = blockIdx.x, tid = threadIdx.x;
    for (int k = tid; k < T_; k += 256) ssc[k] = score[b * T_ + k];
    __syncthreads();
    for (int round = 0; round < BEAM_; ++round) {
        float best = -__builtin_inff(); int bi = 0;
        for (int k = tid; k < T_; k += 256) {
            float v = ssc[k];
            if (v > best) { best = v; bi = k; }
        }
        rv[tid] = best; ri[tid] = bi;
        __syncthreads();
        for (int off = 128; off > 0; off >>= 1) {
            if (tid < off && rv[tid + off] > rv[tid]) {
                rv[tid] = rv[tid + off]; ri[tid] = ri[tid + off];
            }
            __syncthreads();
        }
        if (round == 0) {
            if (tid == 0) { smax = rv[0]; mb[b] = rv[0]; }
            __syncthreads();
            float mx = smax;
            for (int k = tid; k < T_; k += 256)
                q[b * T_ + k] = f2bf(__expf(ssc[k] - mx));
            __syncthreads();
        }
        if (tid == 0) {
            idx5[b * BEAM_ + round] = ri[0];
            ssc[ri[0]] = -__builtin_inff();
        }
        __syncthreads();
    }
}

// ---------------------------------------------------------------------------
// Kernel 5 (per step): score[b,j] = mb[b] + log((Q P^T)[b,j]) + em[b,s,j],
// masked by beam-reachability. 128 blocks (one 16-col tile each),
// 8 waves split K=2048 and reduce partial WMMA accumulators through LDS.
// ---------------------------------------------------------------------------
__global__ void __launch_bounds__(256) step_kernel(
        const unsigned short* __restrict__ q, const unsigned short* __restrict__ P,
        const float* __restrict__ mb, const int* __restrict__ idx5,
        const float* __restrict__ em, const float* __restrict__ A,
        const unsigned char* __restrict__ mask, float* __restrict__ score, int s) {
    __shared__ float lds[8 * 256];
    int wave = threadIdx.x >> 5;
    int lane = threadIdx.x & 31;
    int jt = blockIdx.x;
    int m  = lane & 15;
    int kh = lane >> 4;
    const unsigned short* qrow = q + m * T_;
    const unsigned short* prow = P + (jt * 16 + m) * T_;   // B col n == lane&15
    v8f acc = {};
    int k0 = wave * 256;
    for (int kk = k0; kk < k0 + 256; kk += 32) {
        v16bf a   = load_frag16(qrow, kk, kh);
        v16bf bfr = load_frag16(prow, kk, kh);
        acc = __builtin_amdgcn_wmma_f32_16x16x32_bf16(
                false, a, false, bfr, (short)0, acc, false, false);
    }
#pragma unroll
    for (int r = 0; r < 8; ++r) lds[wave * 256 + r * 32 + lane] = acc[r];
    __syncthreads();
    int t = threadIdx.x;
    float tot = 0.f;
#pragma unroll
    for (int w = 0; w < 8; ++w) tot += lds[w * 256 + t];
    int r = t >> 5, l = t & 31;
    int M = r + 8 * (l >> 4);
    int N = l & 15;
    if (M < B_) {                               // rows 8..15 are padding
        int b = M, j = jt * 16 + N;
        float s5 = 0.f;
#pragma unroll
        for (int i = 0; i < BEAM_; ++i) s5 += A[idx5[b * BEAM_ + i] * T_ + j];
        float ns = mb[b] + __logf(tot) + em[b * S_ * T_ + s * T_ + j];
        if (s5 == 0.f) ns = -__builtin_inff();
        if (mask[b * S_ + s]) score[b * T_ + j] = ns;
    }
}

// ---------------------------------------------------------------------------
// Kernel 6: denominator via top-5 logsumexp + log(T/BEAM), combine to scalar.
// ---------------------------------------------------------------------------
__global__ void __launch_bounds__(256) final_kernel(
        const float* __restrict__ score, const float* __restrict__ num,
        const unsigned char* __restrict__ mask, float* __restrict__ out) {
    __shared__ float ssc[T_];
    __shared__ float rv[256];
    __shared__ int   ri[256];
    __shared__ float top[BEAM_];
    int tid = threadIdx.x;
    float llh = 0.f;
    for (int b = 0; b < B_; ++b) {
        for (int k = tid; k < T_; k += 256) ssc[k] = score[b * T_ + k];
        __syncthreads();
        for (int round = 0; round < BEAM_; ++round) {
            float best = -__builtin_inff(); int bi = 0;
            for (int k = tid; k < T_; k += 256) {
                float v = ssc[k];
                if (v > best) { best = v; bi = k; }
            }
            rv[tid] = best; ri[tid] = bi;
            __syncthreads();
            for (int off = 128; off > 0; off >>= 1) {
                if (tid < off && rv[tid + off] > rv[tid]) {
                    rv[tid] = rv[tid + off]; ri[tid] = ri[tid + off];
                }
                __syncthreads();
            }
            if (tid == 0) { top[round] = rv[0]; ssc[ri[0]] = -__builtin_inff(); }
            __syncthreads();
        }
        if (tid == 0) {
            float mx = top[0], sum = 0.f;
            for (int i = 0; i < BEAM_; ++i) sum += __expf(top[i] - mx);
            float den = mx + __logf(sum) + __logf((float)T_ / (float)BEAM_);
            llh += num[b] - den;
        }
        __syncthreads();
    }
    if (tid == 0) {
        float mf = 0.f;
        for (int i = 0; i < B_ * S_; ++i) mf += mask[i] ? 1.f : 0.f;
        out[0] = llh / mf;
    }
}

// ---------------------------------------------------------------------------
extern "C" void kernel_launch(void* const* d_in, const int* in_sizes, int n_in,
                              void* d_out, int out_size, void* d_ws, size_t ws_size,
                              hipStream_t stream) {
    (void)in_sizes; (void)n_in; (void)out_size; (void)ws_size;
    const float* emissions      = (const float*)d_in[0];         // (8,32,2048)
    const int* tags             = (const int*)d_in[1];           // (8,32)
    const float* E              = (const float*)d_in[2];         // (2048,256)
    const float* A              = (const float*)d_in[3];         // (2048,2048)
    const unsigned char* mask   = (const unsigned char*)d_in[4]; // (8,32) bool

    char* ws = (char*)d_ws;
    size_t off = 0;
    unsigned short* P   = (unsigned short*)(ws + off); off += (size_t)T_ * T_ * 2; // 8 MB bf16
    unsigned short* Ebf = (unsigned short*)(ws + off); off += (size_t)T_ * D_ * 2; // 1 MB bf16
    float* score        = (float*)(ws + off);          off += (size_t)B_ * T_ * 4; // 64 KB
    unsigned short* q   = (unsigned short*)(ws + off); off += (size_t)16 * T_ * 2; // 64 KB bf16
    float* mb           = (float*)(ws + off);          off += 256;
    int* idx5           = (int*)(ws + off);            off += 256;
    float* num          = (float*)(ws + off);          off += 256;
    float* out          = (float*)d_out;

    // E -> bf16 copy (enables pure b128 fragment loads in the GEMM)
    ebf_kernel<<<(T_ * D_) / 256, 256, 0, stream>>>(E, Ebf);
    // P = exp(relu(A .* (E E^T)))  — WMMA bf16 GEMM, 16384 tiles / 8 waves per block
    trans_exp_kernel<<<2048, 256, 0, stream>>>(Ebf, A, P);
    // score <- emissions[:,0,:]; zero Q padding rows
    init_kernel<<<64, 256, 0, stream>>>(emissions, score, q);
    // numerator per batch
    numer_kernel<<<8, 256, 0, stream>>>(emissions, tags, E, A, mask, num);

    // sequential beam-pruned forward scan: each step = bf16 WMMA GEMM in exp-space
    for (int s = 1; s < S_; ++s) {
        prep_kernel<<<8, 256, 0, stream>>>(score, mb, idx5, q);
        step_kernel<<<128, 256, 0, stream>>>(q, P, mb, idx5, emissions, A, mask, score, s);
    }

    final_kernel<<<1, 256, 0, stream>>>(score, num, mask, out);
}
